// BNN_MNIST_39058432590102
// MI455X (gfx1250) — 8-bit matrix operand order repaired and checked by reading, compiled, not run on hardware
//
#include <hip/hip_runtime.h>

typedef int v8i __attribute__((ext_vector_type(8)));
typedef int v4i __attribute__((ext_vector_type(4)));
typedef int v2i __attribute__((ext_vector_type(2)));

#define BNN_EPS 1e-5f

__global__ void k_bin_w2(const float* __restrict__ w2, signed char* __restrict__ w2b) {
    int idx = blockIdx.x * blockDim.x + threadIdx.x;
    int tap = idx % 9;
    int oc  = idx / 9;
    w2b[tap * 4096 + oc] = (w2[idx] >= 0.f) ? 1 : -1;
}

__global__ void k_bin_wfc1(const float* __restrict__ wfc1, signed char* __restrict__ w1b) {
    int idx = blockIdx.x * blockDim.x + threadIdx.x;
    w1b[idx] = (wfc1[idx] >= 0.f) ? 1 : -1;
}

__global__ void k_conv1_pool_bn_bin(
    const float* __restrict__ x,  const float* __restrict__ w1,
    const float* __restrict__ b1, const float* __restrict__ g1,
    const float* __restrict__ be1,const float* __restrict__ m1,
    const float* __restrict__ v1, signed char* __restrict__ act1)
{
    int idx  = blockIdx.x * blockDim.x + threadIdx.x;
    int c    = idx & 63;
    int rest = idx >> 6;
    int pix  = rest % 196;
    int b    = rest / 196;
    int py = pix / 14, px = pix % 14;

    float s[9];
#pragma unroll
    for (int j = 0; j < 9; ++j) s[j] = (w1[c * 9 + j] >= 0.f) ? 1.f : -1.f;

    const float* xb = x + (long)b * 784;
    float mx = -3.4e38f;
#pragma unroll
    for (int dy = 0; dy < 2; ++dy)
#pragma unroll
    for (int dx = 0; dx < 2; ++dx) {
        int y = 2 * py + dy, xx = 2 * px + dx;
        float acc = 0.f;
#pragma unroll
        for (int ky = 0; ky < 3; ++ky) {
            int iy = y + ky - 1;
            if (iy < 0 || iy >= 28) continue;
#pragma unroll
            for (int kx = 0; kx < 3; ++kx) {
                int ix = xx + kx - 1;
                if (ix < 0 || ix >= 28) continue;
                acc += xb[iy * 28 + ix] * s[ky * 3 + kx];
            }
        }
        mx = fmaxf(mx, acc);
    }
    float h   = mx + b1[c];
    float inv = rsqrtf(v1[c] + BNN_EPS);
    float vbn = (h - m1[c]) * (g1[c] * inv) + be1[c];
    act1[((long)b * 196 + pix) * 64 + c] = (vbn >= 0.f) ? 1 : -1;
}

__global__ void __launch_bounds__(128)
k_conv2_wmma(const signed char* __restrict__ act1,
             const signed char* __restrict__ w2b,
             short* __restrict__ c2raw)
{
    int b    = blockIdx.x;
    int tid  = threadIdx.x;
    int wave = tid >> 5;
    int lane = tid & 31;
    int row16 = lane & 15;
    int kh   = lane >> 4;
    int kb   = kh * 8;

    __shared__ signed char Bt[16 * 64] __attribute__((aligned(16)));

    v8i afrag[9];
#pragma unroll
    for (int t = 0; t < 9; ++t) {
        const signed char* ab = w2b + t * 4096 + (wave * 16 + row16) * 64 + kb;
        v2i q0 = *(const v2i*)(ab);
        v2i q1 = *(const v2i*)(ab + 16);
        v2i q2 = *(const v2i*)(ab + 32);
        v2i q3 = *(const v2i*)(ab + 48);
        v8i a = { q0.x, q0.y, q1.x, q1.y, q2.x, q2.y, q3.x, q3.y };
        afrag[t] = a;
    }

    const signed char* a1b = act1 + (long)b * 196 * 64;
    int bpix = tid >> 3;
    int bseg = tid & 7;

    for (int nt = 0; nt < 13; ++nt) {
        v8i acc = { 0, 0, 0, 0, 0, 0, 0, 0 };
#pragma unroll
        for (int t = 0; t < 9; ++t) {
            const int ky = t / 3, kx = t % 3;
            int p = nt * 16 + bpix;
            long val = 0;
            if (p < 196) {
                int y = p / 14, xx = p % 14;
                int iy = y + ky - 1, ix = xx + kx - 1;
                if (iy >= 0 && iy < 14 && ix >= 0 && ix < 14)
                    val = *(const long*)(a1b + (iy * 14 + ix) * 64 + bseg * 8);
            }
            *(long*)(&Bt[bpix * 64 + bseg * 8]) = val;
            __syncthreads();

            const signed char* bb = &Bt[(lane & 15) * 64 + kh * 8];
            v2i r0 = *(const v2i*)(bb);
            v2i r1 = *(const v2i*)(bb + 16);
            v2i r2 = *(const v2i*)(bb + 32);
            v2i r3 = *(const v2i*)(bb + 48);
            v8i bfrag = { r0.x, r0.y, r1.x, r1.y, r2.x, r2.y, r3.x, r3.y };
            acc = __builtin_amdgcn_wmma_i32_16x16x64_iu8(
                true, afrag[t], true, bfrag, acc, false, false);
            __syncthreads();
        }
        int p = nt * 16 + (lane & 15);
        if (p < 196) {
            short* dst = c2raw + ((long)b * 64 + wave * 16 + 8 * kh) * 196 + p;
#pragma unroll
            for (int r = 0; r < 8; ++r)
                dst[(long)r * 196] = (short)acc[r];
        }
    }
}

__global__ void k_pool2_bn_bin(const short* __restrict__ c2raw,
                               const float* __restrict__ b2,
                               const float* __restrict__ g2,
                               const float* __restrict__ be2,
                               const float* __restrict__ m2,
                               const float* __restrict__ v2,
                               signed char* __restrict__ act2)
{
    int idx = blockIdx.x * blockDim.x + threadIdx.x;
    int pp = idx % 49;
    int c  = (idx / 49) % 64;
    int b  = idx / (49 * 64);
    int py = pp / 7, px = pp % 7;
    const short* src = c2raw + ((long)b * 64 + c) * 196;
    int y = 2 * py, xx = 2 * px;
    int m = src[y * 14 + xx];
    int t;
    t = src[y * 14 + xx + 1];       if (t > m) m = t;
    t = src[(y + 1) * 14 + xx];     if (t > m) m = t;
    t = src[(y + 1) * 14 + xx + 1]; if (t > m) m = t;
    float h   = (float)m + b2[c];
    float inv = rsqrtf(v2[c] + BNN_EPS);
    float vbn = (h - m2[c]) * (g2[c] * inv) + be2[c];
    act2[(long)b * 3136 + c * 49 + pp] = (vbn >= 0.f) ? 1 : -1;
}

__global__ void __launch_bounds__(256)
k_fc1_wmma(const signed char* __restrict__ act2,
           const signed char* __restrict__ w1b,
           const float* __restrict__ bfc1,
           const float* __restrict__ g3,
           const float* __restrict__ be3,
           const float* __restrict__ m3,
           const float* __restrict__ v3,
           signed char* __restrict__ act3)
{
    int ntile  = blockIdx.x >> 4;
    int mgroup = blockIdx.x & 15;
    int tid  = threadIdx.x;
    int wave = tid >> 5;
    int lane = tid & 31;
    int mtile = mgroup * 8 + wave;
    int kh = lane >> 4, kb = kh * 8;
    int row16 = lane & 15;

    __shared__ signed char Bt[64 * 16] __attribute__((aligned(16)));

    int bcol = tid >> 4;
    int bseg = tid & 15;
    const signed char* a2   = act2 + (long)(ntile * 16 + bcol) * 3136 + bseg * 4;
    const signed char* arow = w1b + (long)(mtile * 16 + row16) * 3136 + kb;

    v8i acc = { 0, 0, 0, 0, 0, 0, 0, 0 };
    for (int k0 = 0; k0 < 49; ++k0) {
        *(int*)(&Bt[bcol * 64 + bseg * 4]) = *(const int*)(a2 + k0 * 64);
        __syncthreads();

        const signed char* ab = arow + k0 * 64;
        v2i q0 = *(const v2i*)(ab);
        v2i q1 = *(const v2i*)(ab + 16);
        v2i q2 = *(const v2i*)(ab + 32);
        v2i q3 = *(const v2i*)(ab + 48);
        v8i afrag = { q0.x, q0.y, q1.x, q1.y, q2.x, q2.y, q3.x, q3.y };

        const signed char* bbp = &Bt[(lane & 15) * 64 + kh * 8];
        v2i r0 = *(const v2i*)(bbp);
        v2i r1 = *(const v2i*)(bbp + 16);
        v2i r2 = *(const v2i*)(bbp + 32);
        v2i r3 = *(const v2i*)(bbp + 48);
        v8i bfrag = { r0.x, r0.y, r1.x, r1.y, r2.x, r2.y, r3.x, r3.y };

        acc = __builtin_amdgcn_wmma_i32_16x16x64_iu8(
            true, afrag, true, bfrag, acc, false, false);
        __syncthreads();
    }

    int bg = ntile * 16 + (lane & 15);
    int m0 = mtile * 16 + 8 * kh;
    long packed = 0;
#pragma unroll
    for (int r = 0; r < 8; ++r) {
        int m = m0 + r;
        float h   = (float)acc[r] + bfc1[m];
        float inv = rsqrtf(v3[m] + BNN_EPS);
        float vbn = (h - m3[m]) * (g3[m] * inv) + be3[m];
        signed char sv = (vbn >= 0.f) ? 1 : -1;
        packed |= ((long)(unsigned char)sv) << (8 * r);
    }
    *(long*)(act3 + (long)bg * 2048 + m0) = packed;
}

__global__ void k_fc2(const signed char* __restrict__ act3,
                      const float* __restrict__ wfc2,
                      const float* __restrict__ bfc2,
                      const float* __restrict__ scale,
                      float* __restrict__ out)
{
    int idx = blockIdx.x * blockDim.x + threadIdx.x;
    if (idx >= 1024 * 10) return;
    int o = idx % 10;
    int b = idx / 10;
    const signed char* a = act3 + (long)b * 2048;
    const float* w = wfc2 + (long)o * 2048;
    int acc = 0;
    for (int k = 0; k < 2048; ++k) {
        int s = (w[k] >= 0.f) ? 1 : -1;
        acc += s * (int)a[k];
    }
    out[idx] = ((float)acc + bfc2[o]) * scale[0];
}

extern "C" void kernel_launch(void* const* d_in, const int* in_sizes, int n_in,
                              void* d_out, int out_size, void* d_ws, size_t ws_size,
                              hipStream_t stream)
{
    const float* x    = (const float*)d_in[0];
    const float* w1   = (const float*)d_in[1];
    const float* b1   = (const float*)d_in[2];
    const float* g1   = (const float*)d_in[3];
    const float* be1  = (const float*)d_in[4];
    const float* m1   = (const float*)d_in[5];
    const float* v1   = (const float*)d_in[6];
    const float* w2   = (const float*)d_in[7];
    const float* b2   = (const float*)d_in[8];
    const float* g2   = (const float*)d_in[9];
    const float* be2  = (const float*)d_in[10];
    const float* m2   = (const float*)d_in[11];
    const float* v2   = (const float*)d_in[12];
    const float* wfc1 = (const float*)d_in[13];
    const float* bfc1 = (const float*)d_in[14];
    const float* g3   = (const float*)d_in[15];
    const float* be3  = (const float*)d_in[16];
    const float* m3   = (const float*)d_in[17];
    const float* v3   = (const float*)d_in[18];
    const float* wfc2 = (const float*)d_in[19];
    const float* bfc2 = (const float*)d_in[20];
    const float* scale= (const float*)d_in[21];
    float* out = (float*)d_out;

    char* ws = (char*)d_ws;
    signed char* act1  = (signed char*)(ws + 0);
    short*       c2raw = (short*)      (ws + 12845056);
    signed char* act2  = (signed char*)(ws + 38535168);
    signed char* act3  = (signed char*)(ws + 41746432);
    signed char* w2b   = (signed char*)(ws + 43843584);
    signed char* w1b   = (signed char*)(ws + 43880448);

    k_bin_w2           <<<144,   256, 0, stream>>>(w2, w2b);
    k_bin_wfc1         <<<25088, 256, 0, stream>>>(wfc1, w1b);
    k_conv1_pool_bn_bin<<<50176, 256, 0, stream>>>(x, w1, b1, g1, be1, m1, v1, act1);
    k_conv2_wmma       <<<1024,  128, 0, stream>>>(act1, w2b, c2raw);
    k_pool2_bn_bin     <<<12544, 256, 0, stream>>>(c2raw, b2, g2, be2, m2, v2, act2);
    k_fc1_wmma         <<<1024,  256, 0, stream>>>(act2, w1b, bfc1, g3, be3, m3, v3, act3);
    k_fc2              <<<40,    256, 0, stream>>>(act3, wfc2, bfc2, scale, out);
}
